// Model_45423574123023
// MI455X (gfx1250) — matrix operand order repaired and checked by reading, not compiled, not run on hardware
//
#include <hip/hip_runtime.h>

// ---------------------------------------------------------------------------
// Types / helpers
// ---------------------------------------------------------------------------
typedef __attribute__((ext_vector_type(16))) __bf16 v16bf;
typedef __attribute__((ext_vector_type(8)))  __bf16 v8bf;
typedef __attribute__((ext_vector_type(8)))  float  v8f;

__device__ __forceinline__ __bf16 f2bf(float x) {
  // round-to-nearest-even f32 -> bf16
  unsigned u = __builtin_bit_cast(unsigned, x);
  u += 0x7FFFu + ((u >> 16) & 1u);
  unsigned short s = (unsigned short)(u >> 16);
  return __builtin_bit_cast(__bf16, s);
}

__device__ __forceinline__ float sigm(float x) { return 1.0f / (1.0f + __expf(-x)); }

#define WMMA_BF16(a, b, c) \
  __builtin_amdgcn_wmma_f32_16x16x32_bf16(false, (a), false, (b), (short)0, (c), false, false)

// Async global->LDS copy of 16 bytes (ASYNCcnt-tracked).  LDS destination is
// the low 32 bits of the generic pointer (generic->LDS truncates, ISA 10.2).
__device__ __forceinline__ void async_copy_b128(void* lds_dst, const void* gsrc) {
  unsigned ldsb = (unsigned)(size_t)lds_dst;
  asm volatile("global_load_async_to_lds_b128 %0, %1, off"
               :: "v"(ldsb), "v"(gsrc) : "memory");
}
__device__ __forceinline__ void wait_async0() {
  asm volatile("s_wait_asynccnt 0x0" ::: "memory");
}

// ---------------------------------------------------------------------------
// utility kernels
// ---------------------------------------------------------------------------
__global__ void zero_kernel(float* __restrict__ p, int n) {
  int i = blockIdx.x * blockDim.x + threadIdx.x;
  if (i < n) p[i] = 0.0f;
}

__global__ void cvt_bf16_kernel(const float* __restrict__ s, __bf16* __restrict__ d, int n) {
  int i = blockIdx.x * blockDim.x + threadIdx.x;
  if (i < n) d[i] = f2bf(s[i]);
}

// convert [N][K] f32 -> [N][Kp] bf16, zero-padded in K
__global__ void cvt_bf16_pad_kernel(const float* __restrict__ s, __bf16* __restrict__ d,
                                    int N, int K, int Kp) {
  int i = blockIdx.x * blockDim.x + threadIdx.x;
  if (i >= N * Kp) return;
  int n = i / Kp, k = i % Kp;
  d[i] = (k < K) ? f2bf(s[(size_t)n * K + k]) : f2bf(0.0f);
}

// ---------------------------------------------------------------------------
// Generic tiled GEMM:  C[M,N] = A[M,K] @ W[N,K]^T (+ b1[n] + b2[n]) (+= C if acc)
// A: f32 global (converted to bf16 on LDS fill).  W: pre-converted bf16 global,
// B tile moved global->LDS with async-to-LDS instructions.
// Requires: N multiple of 64, K multiple of 32 (true for all call sites).
// Block: 256 threads = 8 waves; tile 128x64; wave = 16 rows x 64 cols.
// ---------------------------------------------------------------------------
#define GBM 128
#define GBN 64
__global__ __launch_bounds__(256) void gemm_kernel(
    const float* __restrict__ A, const __bf16* __restrict__ W,
    const float* __restrict__ b1, const float* __restrict__ b2,
    float* __restrict__ C, int M, int N, int K, int acc)
{
  __shared__ __bf16 As[GBM][40];   // 32 cols used; stride 40 halves (80B) for banks
  __shared__ __bf16 Bs[GBN][40];

  const int tid  = threadIdx.x;
  const int lane = tid & 31;
  const int wv   = tid >> 5;
  const int tileN = blockIdx.x * GBN;
  const int tileM = blockIdx.y * GBM;

  v8f accf[4];
#pragma unroll
  for (int nt = 0; nt < 4; ++nt)
#pragma unroll
    for (int r = 0; r < 8; ++r) accf[nt][r] = 0.0f;

  const int row = lane & 15;
  const int hi  = lane >> 4;
  const int kb  = hi * 8;     // A-fragment K sub-base
  const int kbb = hi * 16;    // unused (B fragment also uses kb)

  // async B-tile assignment: one b128 (8 bf16) per thread covers 64x32 exactly
  const int br   = tid >> 2;        // row 0..63
  const int bseg = tid & 3;         // 8-elem segment 0..3

  for (int k0 = 0; k0 < K; k0 += 32) {
    // ---- async copy B tile (bf16) straight into LDS ----
    async_copy_b128(&Bs[br][bseg * 8],
                    W + (size_t)(tileN + br) * K + k0 + bseg * 8);

    // ---- cooperative A tile load (f32 -> bf16) while B copy is in flight ----
    for (int it = tid; it < GBM * 8; it += 256) {
      int r  = it >> 3;
      int c4 = (it & 7) * 4;
      int m  = tileM + r;
      float v0 = 0.f, v1 = 0.f, v2 = 0.f, v3 = 0.f;
      if (m < M) {
        const float* ap = A + (size_t)m * K + k0 + c4;
        float4 f = *(const float4*)ap;
        v0 = f.x; v1 = f.y; v2 = f.z; v3 = f.w;
      }
      As[r][c4 + 0] = f2bf(v0); As[r][c4 + 1] = f2bf(v1);
      As[r][c4 + 2] = f2bf(v2); As[r][c4 + 3] = f2bf(v3);
    }
    wait_async0();
    __syncthreads();

    // ---- fragments + WMMA ----
    const int row_in = wv * 16 + row;
    v16bf a;
#pragma unroll
    for (int j = 0; j < 8; ++j) {
      a[j]     = As[row_in][kb + j];
      a[8 + j] = As[row_in][16 + kb + j];
    }
#pragma unroll
    for (int nt = 0; nt < 4; ++nt) {
      v16bf bf;
#pragma unroll
      for (int e = 0; e < 8; ++e) { bf[e] = Bs[nt * 16 + row][kb + e]; bf[8 + e] = Bs[nt * 16 + row][16 + kb + e]; }
      accf[nt] = WMMA_BF16(a, bf, accf[nt]);
    }
    __syncthreads();
  }

  // ---- epilogue (C fragment: m = r + 8*hi, n = lane&15) ----
  const int mbase = tileM + wv * 16 + hi * 8;
#pragma unroll
  for (int nt = 0; nt < 4; ++nt) {
    int n = tileN + nt * 16 + row;
#pragma unroll
    for (int r = 0; r < 8; ++r) {
      int m = mbase + r;
      if (m < M) {
        float v = accf[nt][r];
        if (b1) v += b1[n];
        if (b2) v += b2[n];
        size_t o = (size_t)m * N + n;
        if (acc) v += C[o];
        C[o] = v;
      }
    }
  }
}

// ---------------------------------------------------------------------------
// Frame-encoder LSTM cell (batch 1920, hidden 512). g holds i|f|g|o gates.
// ---------------------------------------------------------------------------
__global__ void enc_cell(const float* __restrict__ g, float* __restrict__ c,
                         float* __restrict__ h, float* __restrict__ ctx,
                         const int* __restrict__ xlen, int t)
{
  int idx = blockIdx.x * blockDim.x + threadIdx.x;
  if (idx >= 1920 * 512) return;
  int n = idx >> 9, j = idx & 511;
  const float* gr = g + (size_t)n * 2048;
  float ig = sigm(gr[j]);
  float fg = sigm(gr[512 + j]);
  float gg = tanhf(gr[1024 + j]);
  float og = sigm(gr[1536 + j]);
  float cn = fg * c[idx] + ig * gg;
  float hn = og * tanhf(cn);
  c[idx] = cn;
  h[idx] = hn;
  if (t == xlen[n] - 1) {
    ctx[(size_t)n * 1024 + j]       = hn;
    ctx[(size_t)n * 1024 + 512 + j] = cn;
  }
}

// ---------------------------------------------------------------------------
// Persistent BiLSTM recurrence. grid.x = 2 (dir 0 fwd, 1 bwd), one WG each.
// Per step: g = xp[t] + h @ Whh^T via WMMA (M=4->16, N=2048, K=512, bf16 Whh),
// then the LSTM cell; h in LDS, c in registers.
// ---------------------------------------------------------------------------
__global__ __launch_bounds__(256) void penc_lstm(
    const float* __restrict__ xp_f, const float* __restrict__ xp_b,
    const __bf16* __restrict__ Whh_f, const __bf16* __restrict__ Whh_b,
    float* __restrict__ out, float* __restrict__ hid0, int T)
{
  __shared__ float h_s[4][528];
  __shared__ float g_s[4][2048];

  const int dir = blockIdx.x;
  const float*  xp  = dir ? xp_b  : xp_f;
  const __bf16* Whh = dir ? Whh_b : Whh_f;

  const int tid  = threadIdx.x;
  const int lane = tid & 31;
  const int wv   = tid >> 5;
  const int row  = lane & 15;
  const int hi   = lane >> 4;
  const int kb   = hi * 8;
  const int kbb  = hi * 16;
  const bool vr  = row < 4;

  for (int it = tid; it < 4 * 528; it += 256) ((float*)h_s)[it] = 0.0f;
  float c_reg[8];
#pragma unroll
  for (int i = 0; i < 8; ++i) c_reg[i] = 0.0f;
  __syncthreads();

  for (int s = 0; s < T; ++s) {
    const int t_eff = dir ? (T - 1 - s) : s;

    v8f accf[16];
#pragma unroll
    for (int nt = 0; nt < 16; ++nt) {
      int n = wv * 256 + nt * 16 + row;
#pragma unroll
      for (int r = 0; r < 8; ++r) {
        int m = r + hi * 8;
        accf[nt][r] = (m < 4) ? xp[((size_t)m * T + t_eff) * 2048 + n] : 0.0f;
      }
    }
    for (int kc = 0; kc < 16; ++kc) {
      v16bf a;
      const float* hrow = &h_s[row & 3][kc * 32];
#pragma unroll
      for (int j = 0; j < 8; ++j) {
        a[j]     = f2bf(vr ? hrow[kb + j]      : 0.0f);
        a[8 + j] = f2bf(vr ? hrow[16 + kb + j] : 0.0f);
      }
#pragma unroll
      for (int nt = 0; nt < 16; ++nt) {
        int n = wv * 256 + nt * 16 + row;
        const __bf16* wr = Whh + (size_t)n * 512 + kc * 32 + kb;
        v8bf blo = *(const v8bf*)wr;
        v8bf bhi = *(const v8bf*)(wr + 16);
        v16bf bf;
#pragma unroll
        for (int e = 0; e < 8; ++e) { bf[e] = blo[e]; bf[8 + e] = bhi[e]; }
        accf[nt] = WMMA_BF16(a, bf, accf[nt]);
      }
    }
#pragma unroll
    for (int nt = 0; nt < 16; ++nt) {
      int n = wv * 256 + nt * 16 + row;
#pragma unroll
      for (int r = 0; r < 8; ++r) {
        int m = r + hi * 8;
        if (m < 4) g_s[m][n] = accf[nt][r];
      }
    }
    __syncthreads();

#pragma unroll
    for (int it = 0; it < 8; ++it) {
      int idx = tid + it * 256;
      int b = idx >> 9, j = idx & 511;
      float ig = sigm(g_s[b][j]);
      float fg = sigm(g_s[b][512 + j]);
      float gg = tanhf(g_s[b][1024 + j]);
      float og = sigm(g_s[b][1536 + j]);
      float cn = fg * c_reg[it] + ig * gg;
      float hn = og * tanhf(cn);
      c_reg[it] = cn;
      h_s[b][j] = hn;
      out[((size_t)b * T + t_eff) * 1024 + dir * 512 + j] = hn;
      if (s == T - 1) hid0[b * 1024 + dir * 512 + j] = hn;
    }
    __syncthreads();
  }
}

// ---------------------------------------------------------------------------
// Persistent attention decoder: single workgroup, 32 steps.
// comb_W padded to K=1056 (bf16), gru weights bf16.
// ---------------------------------------------------------------------------
__global__ __launch_bounds__(256) void decoder_kernel(
    const float* __restrict__ enc_out, const float* __restrict__ hid0,
    const float* __restrict__ attn_W, const float* __restrict__ attn_b,
    const __bf16* __restrict__ comb_Wb, const float* __restrict__ comb_b,
    const __bf16* __restrict__ gWihb, const __bf16* __restrict__ gWhhb,
    const float* __restrict__ gbih, const float* __restrict__ gbhh,
    const float* __restrict__ oW, const float* __restrict__ ob,
    float* __restrict__ gi_g, float* __restrict__ gh_g,
    float* __restrict__ dout)
{
  __shared__ float h_s[4][1040];    // cols 0..1023
  __shared__ float cat_s[4][1056];  // [inp(8) | h or ctxv(1024)]; tail zeroed
  __shared__ float x_s[4][520];     // cols 0..511
  __shared__ float sc_s[4][32];
  __shared__ float inp_s[4][8];

  const int tid  = threadIdx.x;
  const int lane = tid & 31;
  const int wv   = tid >> 5;
  const int row  = lane & 15;
  const int hi   = lane >> 4;
  const int kb   = hi * 8;
  const int kbb  = hi * 16;
  const bool vr  = row < 4;

  for (int it = tid; it < 32; it += 256) ((float*)inp_s)[it] = 0.0f;
  for (int it = tid; it < 4096; it += 256) h_s[it >> 10][it & 1023] = hid0[it];
  for (int it = tid; it < 4 * 1056; it += 256) ((float*)cat_s)[it] = 0.0f;
  __syncthreads();

  for (int step = 0; step < 32; ++step) {
    // 1. cat = [inp | h]
    for (int it = tid; it < 4 * 1032; it += 256) {
      int b = it / 1032, k = it % 1032;
      cat_s[b][k] = (k < 8) ? inp_s[b][k] : h_s[b][k - 8];
    }
    __syncthreads();

    // 2. attention scores + softmax
    if (tid < 120) {
      int b = tid / 30, a2 = tid % 30;
      const float* wr = attn_W + (size_t)a2 * 1032;
      float sum = attn_b[a2];
      for (int k = 0; k < 1032; ++k) sum += cat_s[b][k] * wr[k];
      sc_s[b][a2] = sum;
    }
    __syncthreads();
    if (tid < 4) {
      float mx = -1e30f;
      for (int a2 = 0; a2 < 30; ++a2) mx = fmaxf(mx, sc_s[tid][a2]);
      float se = 0.0f;
      for (int a2 = 0; a2 < 30; ++a2) { float e = __expf(sc_s[tid][a2] - mx); sc_s[tid][a2] = e; se += e; }
      float inv = 1.0f / se;
      for (int a2 = 0; a2 < 30; ++a2) sc_s[tid][a2] *= inv;
    }
    __syncthreads();

    // 3. ctxv = aw @ enc_out -> cat[:, 8:1032]
    for (int it = tid; it < 4096; it += 256) {
      int b = it >> 10, k = it & 1023;
      float v = 0.0f;
      for (int a2 = 0; a2 < 30; ++a2)
        v += sc_s[b][a2] * enc_out[((size_t)b * 30 + a2) * 1024 + k];
      cat_s[b][8 + k] = v;
    }
    __syncthreads();

    // 4. comb: x = relu(cat @ comb_W^T + comb_b)  (M=4->16, N=512, Kp=1056)
    {
      v8f accf[4];
#pragma unroll
      for (int nt = 0; nt < 4; ++nt) {
        int n = wv * 64 + nt * 16 + row;
        float bb = comb_b[n];
#pragma unroll
        for (int r = 0; r < 8; ++r) accf[nt][r] = bb;
      }
      for (int kc = 0; kc < 33; ++kc) {
        v16bf a;
        const float* cr = &cat_s[row & 3][kc * 32];
#pragma unroll
        for (int j = 0; j < 8; ++j) {
          a[j]     = f2bf(vr ? cr[kb + j]      : 0.0f);
          a[8 + j] = f2bf(vr ? cr[16 + kb + j] : 0.0f);
        }
#pragma unroll
        for (int nt = 0; nt < 4; ++nt) {
          int n = wv * 64 + nt * 16 + row;
          const __bf16* wr = comb_Wb + (size_t)n * 1056 + kc * 32 + kb;
          v8bf blo = *(const v8bf*)wr;
          v8bf bhi = *(const v8bf*)(wr + 16);
          v16bf bf;
#pragma unroll
          for (int e = 0; e < 8; ++e) { bf[e] = blo[e]; bf[8 + e] = bhi[e]; }
          accf[nt] = WMMA_BF16(a, bf, accf[nt]);
        }
      }
#pragma unroll
      for (int nt = 0; nt < 4; ++nt) {
        int n = wv * 64 + nt * 16 + row;
#pragma unroll
        for (int r = 0; r < 8; ++r) {
          int m = r + hi * 8;
          if (m < 4) x_s[m][n] = fmaxf(accf[nt][r], 0.0f);
        }
      }
    }
    __syncthreads();

    // 5a. gi = x @ gWih^T + gbih   (N=3072, K=512)
    for (int nt = 0; nt < 24; ++nt) {
      int n = wv * 384 + nt * 16 + row;
      v8f acc;
      float bb = gbih[n];
#pragma unroll
      for (int r = 0; r < 8; ++r) acc[r] = bb;
      for (int kc = 0; kc < 16; ++kc) {
        v16bf a;
        const float* xr = &x_s[row & 3][kc * 32];
#pragma unroll
        for (int j = 0; j < 8; ++j) {
          a[j]     = f2bf(vr ? xr[kb + j]      : 0.0f);
          a[8 + j] = f2bf(vr ? xr[16 + kb + j] : 0.0f);
        }
        const __bf16* wr = gWihb + (size_t)n * 512 + kc * 32 + kb;
        v8bf blo = *(const v8bf*)wr;
        v8bf bhi = *(const v8bf*)(wr + 16);
        v16bf bf;
#pragma unroll
        for (int e = 0; e < 8; ++e) { bf[e] = blo[e]; bf[8 + e] = bhi[e]; }
        acc = WMMA_BF16(a, bf, acc);
      }
#pragma unroll
      for (int r = 0; r < 8; ++r) {
        int m = r + hi * 8;
        if (m < 4) gi_g[m * 3072 + n] = acc[r];
      }
    }
    // 5b. gh = h @ gWhh^T + gbhh   (N=3072, K=1024)
    for (int nt = 0; nt < 24; ++nt) {
      int n = wv * 384 + nt * 16 + row;
      v8f acc;
      float bb = gbhh[n];
#pragma unroll
      for (int r = 0; r < 8; ++r) acc[r] = bb;
      for (int kc = 0; kc < 32; ++kc) {
        v16bf a;
        const float* hr = &h_s[row & 3][kc * 32];
#pragma unroll
        for (int j = 0; j < 8; ++j) {
          a[j]     = f2bf(vr ? hr[kb + j]      : 0.0f);
          a[8 + j] = f2bf(vr ? hr[16 + kb + j] : 0.0f);
        }
        const __bf16* wr = gWhhb + (size_t)n * 1024 + kc * 32 + kb;
        v8bf blo = *(const v8bf*)wr;
        v8bf bhi = *(const v8bf*)(wr + 16);
        v16bf bf;
#pragma unroll
        for (int e = 0; e < 8; ++e) { bf[e] = blo[e]; bf[8 + e] = bhi[e]; }
        acc = WMMA_BF16(a, bf, acc);
      }
#pragma unroll
      for (int r = 0; r < 8; ++r) {
        int m = r + hi * 8;
        if (m < 4) gh_g[m * 3072 + n] = acc[r];
      }
    }
    __threadfence();
    __syncthreads();

    // 6. GRU cell
    for (int it = tid; it < 4096; it += 256) {
      int b = it >> 10, j = it & 1023;
      float r_ = sigm(gi_g[b * 3072 + j]        + gh_g[b * 3072 + j]);
      float z_ = sigm(gi_g[b * 3072 + 1024 + j] + gh_g[b * 3072 + 1024 + j]);
      float n_ = tanhf(gi_g[b * 3072 + 2048 + j] + r_ * gh_g[b * 3072 + 2048 + j]);
      h_s[b][j] = (1.0f - z_) * n_ + z_ * h_s[b][j];
    }
    __syncthreads();

    // 7. logits + log_softmax; feed back
    if (tid < 32) {
      int b = tid >> 3, u = tid & 7;
      const float* wr = oW + (size_t)u * 1024;
      float sum = ob[u];
      for (int k = 0; k < 1024; ++k) sum += h_s[b][k] * wr[k];
      sc_s[b][u] = sum;
    }
    __syncthreads();
    if (tid < 4) {
      float mx = -1e30f;
      for (int u = 0; u < 8; ++u) mx = fmaxf(mx, sc_s[tid][u]);
      float se = 0.0f;
      for (int u = 0; u < 8; ++u) se += __expf(sc_s[tid][u] - mx);
      float lse = mx + __logf(se);
      for (int u = 0; u < 8; ++u) {
        float lp = sc_s[tid][u] - lse;
        inp_s[tid][u] = lp;
        dout[(tid * 32 + step) * 8 + u] = lp;
      }
    }
    __syncthreads();
  }
}

// ---------------------------------------------------------------------------
// Host orchestration
// ---------------------------------------------------------------------------
extern "C" void kernel_launch(void* const* d_in, const int* in_sizes, int n_in,
                              void* d_out, int out_size, void* d_ws, size_t ws_size,
                              hipStream_t stream) {
  (void)in_sizes; (void)n_in; (void)out_size; (void)ws_size;

  const float* xs      = (const float*)d_in[0];
  const int*   xlen    = (const int*)  d_in[1];
  const float* enc_Wih = (const float*)d_in[2];
  const float* enc_Whh = (const float*)d_in[3];
  const float* enc_bih = (const float*)d_in[4];
  const float* enc_bhh = (const float*)d_in[5];
  const float* pWih_f  = (const float*)d_in[6];
  const float* pWhh_f  = (const float*)d_in[7];
  const float* pbih_f  = (const float*)d_in[8];
  const float* pbhh_f  = (const float*)d_in[9];
  const float* pWih_b  = (const float*)d_in[10];
  const float* pWhh_b  = (const float*)d_in[11];
  const float* pbih_b  = (const float*)d_in[12];
  const float* pbhh_b  = (const float*)d_in[13];
  const float* attn_W  = (const float*)d_in[14];
  const float* attn_b  = (const float*)d_in[15];
  const float* comb_W  = (const float*)d_in[16];
  const float* comb_b  = (const float*)d_in[17];
  const float* gWih    = (const float*)d_in[18];
  const float* gWhh    = (const float*)d_in[19];
  const float* gbih    = (const float*)d_in[20];
  const float* gbhh    = (const float*)d_in[21];
  const float* oW      = (const float*)d_in[22];
  const float* ob      = (const float*)d_in[23];

  float* ws = (float*)d_ws;
  size_t off = 0;
  auto alloc = [&](size_t nfloats) { float* p = ws + off; off += nfloats; return p; };

  float* g    = alloc(1920ull * 2048);  // enc gates; later xp_f / xp_b
  float* hC   = alloc(1920ull * 512);   // enc h; later layer out ping
  float* cC   = alloc(1920ull * 512);   // enc c; later layer out pong
  float* ctx  = alloc(1920ull * 1024);  // [h;c] gather -> penc layer-0 input
  float* hid0 = alloc(4096);
  float* gi   = alloc(4 * 3072);
  float* gh   = alloc(4 * 3072);
  // bf16 weight staging (sizes in floats = bf16_elems/2; all 16B aligned)
  __bf16* encWih_bf = (__bf16*)alloc(2048ull * 32 / 2);
  __bf16* encWhh_bf = (__bf16*)alloc(2048ull * 512 / 2);
  __bf16* wbuf_bf   = (__bf16*)alloc(2048ull * 2048 / 2);   // per-layer Wih staging
  __bf16* whhF_bf   = (__bf16*)alloc(2048ull * 512 / 2);
  __bf16* whhB_bf   = (__bf16*)alloc(2048ull * 512 / 2);
  __bf16* comb_bf   = (__bf16*)alloc(512ull * 1056 / 2);
  __bf16* gWih_bf   = (__bf16*)alloc(3072ull * 512 / 2);
  __bf16* gWhh_bf   = (__bf16*)alloc(3072ull * 1024 / 2);

  float* xpf = g;
  float* xpb = g + 960ull * 2048;
  float* oA  = hC;
  float* oB  = cC;

  auto cvt = [&](const float* s, __bf16* d, int n) {
    cvt_bf16_kernel<<<dim3((n + 255) / 256), 256, 0, stream>>>(s, d, n);
  };

  // one-time conversions
  cvt(enc_Wih, encWih_bf, 2048 * 32);
  cvt(enc_Whh, encWhh_bf, 2048 * 512);
  cvt(gWih, gWih_bf, 3072 * 512);
  cvt(gWhh, gWhh_bf, 3072 * 1024);
  cvt_bf16_pad_kernel<<<dim3((512 * 1056 + 255) / 256), 256, 0, stream>>>(
      comb_W, comb_bf, 512, 1032, 1056);

  // zero h and c (contiguous)
  zero_kernel<<<dim3((1920 * 512 * 2 + 255) / 256), 256, 0, stream>>>(hC, 1920 * 512 * 2);

  // ---- frame-level encoder LSTM: 12 steps ----
  for (int t = 0; t < 12; ++t) {
    gemm_kernel<<<dim3(32, 15), 256, 0, stream>>>(
        xs + (size_t)t * 1920 * 32, encWih_bf, enc_bih, enc_bhh, g, 1920, 2048, 32, 0);
    gemm_kernel<<<dim3(32, 15), 256, 0, stream>>>(
        hC, encWhh_bf, nullptr, nullptr, g, 1920, 2048, 512, 1);
    enc_cell<<<dim3(3840), 256, 0, stream>>>(g, cC, hC, ctx, xlen, t);
  }

  // ---- pyramidal BiLSTM stack ----
  const float* X = ctx;
  float* outb = oA;
  for (int i = 0; i < 4; ++i) {
    int T = 240 >> i;
    int M = 4 * T;
    cvt(pWih_f + (size_t)i * 2048 * 2048, wbuf_bf, 2048 * 2048);
    gemm_kernel<<<dim3(32, (M + 127) / 128), 256, 0, stream>>>(
        X, wbuf_bf, pbih_f + i * 2048, pbhh_f + i * 2048, xpf, M, 2048, 2048, 0);
    cvt(pWih_b + (size_t)i * 2048 * 2048, wbuf_bf, 2048 * 2048);
    gemm_kernel<<<dim3(32, (M + 127) / 128), 256, 0, stream>>>(
        X, wbuf_bf, pbih_b + i * 2048, pbhh_b + i * 2048, xpb, M, 2048, 2048, 0);
    cvt(pWhh_f + (size_t)i * 2048 * 512, whhF_bf, 2048 * 512);
    cvt(pWhh_b + (size_t)i * 2048 * 512, whhB_bf, 2048 * 512);
    penc_lstm<<<dim3(2), 256, 0, stream>>>(xpf, xpb, whhF_bf, whhB_bf, outb, hid0, T);
    X = outb;
    outb = (outb == oA) ? oB : oA;
  }
  const float* enc_out = X;   // [4, 30, 1024]

  // ---- attention decoder (persistent, 32 steps) ----
  decoder_kernel<<<dim3(1), 256, 0, stream>>>(
      enc_out, hid0, attn_W, attn_b, comb_bf, comb_b,
      gWih_bf, gWhh_bf, gbih, gbhh, oW, ob, gi, gh, (float*)d_out);
}
